// SuperGlueAttentionalGNN_29927332119006
// MI455X (gfx1250) — compile-verified, matrix operand order checked by reading, not run on hardware
//
#include <hip/hip_runtime.h>
#include <hip/hip_bf16.h>

#define BQ 4
#define CC 256
#define NN 2048
#define HH 4
#define LL 18

typedef __attribute__((ext_vector_type(16))) __bf16 v16bf;
typedef __attribute__((ext_vector_type(8)))  float  v8f;

// Fragment is two contiguous 16-byte runs in K (ISA 7.12.2):
//   u[0..7]  <- K = hi*8 + 0..7
//   u[8..15] <- K = 16 + hi*8 + 0..7
union BF16Frag {
  v16bf v;
  uint4 q[2];
  unsigned short u[16];
};

__device__ __forceinline__ unsigned short f2bf(float f) {
  unsigned int x = __float_as_uint(f);
  x += 0x7FFFu + ((x >> 16) & 1u);   // round-to-nearest-even
  return (unsigned short)(x >> 16);
}

// ---------------------------------------------------------------------------
// Generic GEMM:  Y[b,m,n] = epilogue( sum_k W[m,k] * X[b,k,n] + bias[m] )
//   X rows < K0 from X0, rows >= K0 from X1 (fused concat).
//   MODE 0: +bias   MODE 1: relu(scale*(.)+shift)   MODE 2: + resid
// Block tile 64 rows x 128 cols, 8 waves (4 row-groups x 2 col-halves),
// each wave computes a 16x64 strip; K staged 64 at a time (2 WMMA k-steps
// per barrier pair). LDS is fragment-major: A source Wt[row][K], B source
// Xt[col][K] -> every fragment is two ds_load_b128 per lane.
// ---------------------------------------------------------------------------
template <int MODE>
__global__ __launch_bounds__(256) void gemm_bf16(
    const unsigned short* __restrict__ Wb,   // [M,K] bf16 row-major
    const float* __restrict__ X0,            // [B,K0,N]
    const float* __restrict__ X1,            // [B,K-K0,N] or null
    const float* __restrict__ bias,          // [M]
    const float* __restrict__ scale,         // [M] or null
    const float* __restrict__ shift,         // [M] or null
    const float* __restrict__ resid,         // [B,M,N] or null
    float* __restrict__ Y,                   // [B,M,N]
    int M, int K, int K0)
{
  __shared__ unsigned short Wt[64][72];    // [row][K-slab 64], 144B rows (16B-aligned)
  __shared__ unsigned short Xt[128][72];   // [col][K-slab 64]

  const int tid  = threadIdx.x;
  const int wv   = tid >> 5;
  const int lane = tid & 31;
  const int ln   = lane & 15;
  const int hi   = lane >> 4;
  const int wr   = (wv & 3) * 16;   // wave row offset within 64-row tile
  const int wc   = (wv >> 2) * 64;  // wave col offset within 128-col tile

  const int rowTile = blockIdx.y * 64;
  const int colG    = blockIdx.x * 128;
  const int b       = colG >> 11;          // 2048 cols per batch
  const int n0      = colG & (NN - 1);

  v8f acc[4];
#pragma unroll
  for (int f = 0; f < 4; ++f)
#pragma unroll
    for (int r = 0; r < 8; ++r) acc[f][r] = 0.0f;

  for (int kk = 0; kk < K; kk += 64) {
    { // stage W tile 64x64: two uint4 global loads + two uint4 LDS stores / thread
      int r = tid >> 2, seg = tid & 3;
      const uint4* src = (const uint4*)(Wb + (size_t)(rowTile + r) * K + kk) + seg;
      *(uint4*)&Wt[r][seg * 8]      = src[0];
      *(uint4*)&Wt[r][32 + seg * 8] = src[4];
    }
    { // stage X tile 64x128 transposed -> Xt[col][k]; fp32->bf16, fused concat
      int kr = tid >> 2, seg = tid & 3;   // 64 k-rows, 32 cols per thread
      int c = kk + kr;
      const float* srcp = (c < K0)
          ? X0 + ((size_t)b * K0 + c) * NN + n0 + seg * 32
          : X1 + ((size_t)b * (K - K0) + (c - K0)) * NN + n0 + seg * 32;
      const float4* src4 = (const float4*)srcp;
#pragma unroll
      for (int c4 = 0; c4 < 8; ++c4) {
        float4 vv = src4[c4];
        int col = seg * 32 + c4 * 4;
        Xt[col + 0][kr] = f2bf(vv.x);
        Xt[col + 1][kr] = f2bf(vv.y);
        Xt[col + 2][kr] = f2bf(vv.z);
        Xt[col + 3][kr] = f2bf(vv.w);
      }
    }
    __syncthreads();

#pragma unroll
    for (int k2 = 0; k2 < 64; k2 += 32) {
      BF16Frag a;
      a.q[0] = *(const uint4*)&Wt[wr + ln][k2 + hi * 8];
      a.q[1] = *(const uint4*)&Wt[wr + ln][k2 + 16 + hi * 8];
#pragma unroll
      for (int f = 0; f < 4; ++f) {
        BF16Frag bb;
        const int col = wc + 16 * f + ln;
        bb.q[0] = *(const uint4*)&Xt[col][k2 + hi * 8];
        bb.q[1] = *(const uint4*)&Xt[col][k2 + 16 + hi * 8];
        acc[f] = __builtin_amdgcn_wmma_f32_16x16x32_bf16(false, a.v, false, bb.v,
                                                         (short)0, acc[f], false, false);
      }
    }
    __syncthreads();
  }

  // epilogue: hoist per-row parameters (8 rows per lane)
  float bi[8], sc[8], sh[8];
#pragma unroll
  for (int r = 0; r < 8; ++r) {
    int m = rowTile + wr + r + hi * 8;
    bi[r] = bias[m];
    if (MODE == 1) { sc[r] = scale[m]; sh[r] = shift[m]; }
  }
#pragma unroll
  for (int f = 0; f < 4; ++f) {
    int n = n0 + wc + 16 * f + ln;
#pragma unroll
    for (int r = 0; r < 8; ++r) {
      int m = rowTile + wr + r + hi * 8;
      float y = acc[f][r] + bi[r];
      if (MODE == 1) y = fmaxf(0.0f, y * sc[r] + sh[r]);
      if (MODE == 2) y += resid[((size_t)b * M + m) * NN + n];
      Y[((size_t)b * M + m) * NN + n] = y;
    }
  }
}

// ---------------------------------------------------------------------------
// Flash attention for one (b,h): 128 query rows per block, stream 64-key tiles.
// Head layout: channel c = d*H + h (SuperGlue interleaved heads), d in [0,64).
// Each wave owns 16 query rows end-to-end -> softmax stats never cross waves.
// LDS fragment-major: Qs[n][d] (A of S), Ks[m][d] (B of S), Vs[d][m] (B of O),
// Ps[wave][n][m] (A of O). All fragment reads are ds_load_b128 pairs.
// ---------------------------------------------------------------------------
__global__ __launch_bounds__(256) void flash_attn(
    const float* __restrict__ Qg, const float* __restrict__ Kg,
    const float* __restrict__ Vg, float* __restrict__ Og)
{
  __shared__ unsigned short Qs[128][72];    // [n][d]
  __shared__ unsigned short Ks[64][72];     // [m][d]
  __shared__ unsigned short Vs[64][72];     // [d][m]
  __shared__ unsigned short Ps[8][16][72];  // per-wave [n][m]

  const int tid  = threadIdx.x;
  const int wv   = tid >> 5;
  const int lane = tid & 31;
  const int ln   = lane & 15;
  const int hi   = lane >> 4;

  const int ntile = blockIdx.x * 128;
  const int b     = blockIdx.y >> 2;
  const int h     = blockIdx.y & 3;

  { // load Q tile: global contiguous in n for fixed d; transpose into Qs[n][d]
    int dr = tid >> 2, seg = tid & 3;
    const float4* src = (const float4*)(Qg + ((size_t)b * CC + dr * HH + h) * NN + ntile + seg * 32);
#pragma unroll
    for (int c4 = 0; c4 < 8; ++c4) {
      float4 vv = src[c4];
      int n = seg * 32 + c4 * 4;
      Qs[n + 0][dr] = f2bf(vv.x);
      Qs[n + 1][dr] = f2bf(vv.y);
      Qs[n + 2][dr] = f2bf(vv.z);
      Qs[n + 3][dr] = f2bf(vv.w);
    }
  }

  v8f oacc[4];
  float rm[8], rs[8];
#pragma unroll
  for (int f = 0; f < 4; ++f)
#pragma unroll
    for (int r = 0; r < 8; ++r) oacc[f][r] = 0.0f;
#pragma unroll
  for (int r = 0; r < 8; ++r) { rm[r] = -1e30f; rs[r] = 0.0f; }

  for (int m0 = 0; m0 < NN; m0 += 64) {
    __syncthreads();
    { // stage K (transposed scatter) and V (packed contiguous) tiles
      int dr = tid >> 2, seg = tid & 3;
      const float4* ks = (const float4*)(Kg + ((size_t)b * CC + dr * HH + h) * NN + m0 + seg * 16);
      const float4* vs = (const float4*)(Vg + ((size_t)b * CC + dr * HH + h) * NN + m0 + seg * 16);
#pragma unroll
      for (int c4 = 0; c4 < 4; ++c4) {
        float4 kv = ks[c4];
        int m = seg * 16 + c4 * 4;
        Ks[m + 0][dr] = f2bf(kv.x);
        Ks[m + 1][dr] = f2bf(kv.y);
        Ks[m + 2][dr] = f2bf(kv.z);
        Ks[m + 3][dr] = f2bf(kv.w);
      }
      union { uint4 q[2]; unsigned short u[16]; } pk;
#pragma unroll
      for (int c4 = 0; c4 < 4; ++c4) {
        float4 vv = vs[c4];
        pk.u[c4 * 4 + 0] = f2bf(vv.x);
        pk.u[c4 * 4 + 1] = f2bf(vv.y);
        pk.u[c4 * 4 + 2] = f2bf(vv.z);
        pk.u[c4 * 4 + 3] = f2bf(vv.w);
      }
      *(uint4*)&Vs[dr][seg * 16 + 0] = pk.q[0];
      *(uint4*)&Vs[dr][seg * 16 + 8] = pk.q[1];
    }
    __syncthreads();

    // S = Q^T K  (rows n, cols m, K-dim d = 64)
    v8f sacc[4];
#pragma unroll
    for (int f = 0; f < 4; ++f)
#pragma unroll
      for (int r = 0; r < 8; ++r) sacc[f][r] = 0.0f;

#pragma unroll
    for (int kk = 0; kk < 64; kk += 32) {
      BF16Frag a;
      a.q[0] = *(const uint4*)&Qs[wv * 16 + ln][kk + hi * 8];
      a.q[1] = *(const uint4*)&Qs[wv * 16 + ln][kk + 16 + hi * 8];
#pragma unroll
      for (int f = 0; f < 4; ++f) {
        BF16Frag bb;
        bb.q[0] = *(const uint4*)&Ks[16 * f + ln][kk + hi * 8];
        bb.q[1] = *(const uint4*)&Ks[16 * f + ln][kk + 16 + hi * 8];
        sacc[f] = __builtin_amdgcn_wmma_f32_16x16x32_bf16(false, a.v, false, bb.v,
                                                          (short)0, sacc[f], false, false);
      }
    }

    // online softmax; row (wv*16 + r + hi*8) lives on 16 lanes of one half-wave
#pragma unroll
    for (int r = 0; r < 8; ++r) {
      float tmax = -1e30f;
#pragma unroll
      for (int f = 0; f < 4; ++f) {
        sacc[f][r] *= 0.125f;                    // 1/sqrt(64)
        tmax = fmaxf(tmax, sacc[f][r]);
      }
#pragma unroll
      for (int msk = 1; msk < 16; msk <<= 1)
        tmax = fmaxf(tmax, __shfl_xor(tmax, msk, 32));
      float mnew  = fmaxf(rm[r], tmax);
      float alpha = __expf(rm[r] - mnew);
      float psum  = 0.0f;
#pragma unroll
      for (int f = 0; f < 4; ++f) {
        float p = __expf(sacc[f][r] - mnew);
        psum += p;
        Ps[wv][r + hi * 8][16 * f + ln] = f2bf(p);
      }
#pragma unroll
      for (int msk = 1; msk < 16; msk <<= 1)
        psum += __shfl_xor(psum, msk, 32);
      rs[r] = rs[r] * alpha + psum;
      rm[r] = mnew;
#pragma unroll
      for (int f = 0; f < 4; ++f) oacc[f][r] *= alpha;
    }

    // O += P V^T  (rows n, cols d, K-dim m = 64); Ps is wave-private
#pragma unroll
    for (int kk = 0; kk < 64; kk += 32) {
      BF16Frag a;
      a.q[0] = *(const uint4*)&Ps[wv][ln][kk + hi * 8];
      a.q[1] = *(const uint4*)&Ps[wv][ln][kk + 16 + hi * 8];
#pragma unroll
      for (int f = 0; f < 4; ++f) {
        BF16Frag bb;
        bb.q[0] = *(const uint4*)&Vs[16 * f + ln][kk + hi * 8];
        bb.q[1] = *(const uint4*)&Vs[16 * f + ln][kk + 16 + hi * 8];
        oacc[f] = __builtin_amdgcn_wmma_f32_16x16x32_bf16(false, a.v, false, bb.v,
                                                          (short)0, oacc[f], false, false);
      }
    }
  }

#pragma unroll
  for (int f = 0; f < 4; ++f) {
    int d = 16 * f + ln;
#pragma unroll
    for (int r = 0; r < 8; ++r) {
      int n = ntile + wv * 16 + r + hi * 8;
      Og[((size_t)b * CC + d * HH + h) * NN + n] = oacc[f][r] / rs[r];
    }
  }
}

__global__ __launch_bounds__(256) void cvt_bf16(const float* __restrict__ in,
                                                unsigned short* __restrict__ out, int n) {
  int i = blockIdx.x * 256 + threadIdx.x;
  if (i < n) out[i] = f2bf(in[i]);
}

// ---------------------------------------------------------------------------
extern "C" void kernel_launch(void* const* d_in, const int* in_sizes, int n_in,
                              void* d_out, int out_size, void* d_ws, size_t ws_size,
                              hipStream_t stream) {
  const float* din0 = (const float*)d_in[0];
  const float* din1 = (const float*)d_in[1];
  const float* Wq = (const float*)d_in[2];  const float* bq = (const float*)d_in[3];
  const float* Wk = (const float*)d_in[4];  const float* bk = (const float*)d_in[5];
  const float* Wv = (const float*)d_in[6];  const float* bv = (const float*)d_in[7];
  const float* Wm = (const float*)d_in[8];  const float* bm = (const float*)d_in[9];
  const float* W1 = (const float*)d_in[10]; const float* b1 = (const float*)d_in[11];
  const float* s1 = (const float*)d_in[12]; const float* t1 = (const float*)d_in[13];
  const float* W2 = (const float*)d_in[14]; const float* b2 = (const float*)d_in[15];
  (void)in_sizes; (void)n_in; (void)out_size; (void)ws_size;

  char* wsp = (char*)d_ws;
  size_t off = 0;
  auto carve = [&](size_t bytes) -> void* {
    void* p = wsp + off;
    off += (bytes + 255) & ~(size_t)255;
    return p;
  };

  const size_t setElems = (size_t)BQ * CC * NN;          // 2M floats per descriptor set
  unsigned short* bWq = (unsigned short*)carve((size_t)LL * CC * CC * 2);
  unsigned short* bWk = (unsigned short*)carve((size_t)LL * CC * CC * 2);
  unsigned short* bWv = (unsigned short*)carve((size_t)LL * CC * CC * 2);
  unsigned short* bWm = (unsigned short*)carve((size_t)LL * CC * CC * 2);
  unsigned short* bW1 = (unsigned short*)carve((size_t)LL * 512 * 512 * 2);
  unsigned short* bW2 = (unsigned short*)carve((size_t)LL * 256 * 512 * 2);
  float* Qb  = (float*)carve(setElems * 4);
  float* Kb  = (float*)carve(setElems * 4);
  float* Vb  = (float*)carve(setElems * 4);
  float* AO  = (float*)carve(setElems * 4);              // attention output (pre-merge)
  float* AM  = (float*)carve(setElems * 4);              // merged attention
  float* Hb  = (float*)carve((size_t)BQ * 512 * NN * 4); // MLP hidden
  float* pA0 = (float*)carve(setElems * 4);
  float* pA1 = (float*)carve(setElems * 4);
  float* pB0 = (float*)carve(setElems * 4);
  float* pB1 = (float*)carve(setElems * 4);

  auto cvt = [&](const float* src, unsigned short* dst, size_t n) {
    cvt_bf16<<<(unsigned)((n + 255) / 256), 256, 0, stream>>>(src, dst, (int)n);
  };
  cvt(Wq, bWq, (size_t)LL * CC * CC);
  cvt(Wk, bWk, (size_t)LL * CC * CC);
  cvt(Wv, bWv, (size_t)LL * CC * CC);
  cvt(Wm, bWm, (size_t)LL * CC * CC);
  cvt(W1, bW1, (size_t)LL * 512 * 512);
  cvt(W2, bW2, (size_t)LL * 256 * 512);

  const dim3 blk(256);
  const dim3 gC(64, 4);   // M=256 GEMMs
  const dim3 gH(64, 8);   // M=512 GEMM
  const dim3 gA(NN / 128, BQ * HH);

  const float* cur0 = din0;
  const float* cur1 = din1;

  for (int l = 0; l < LL; ++l) {
    float *n0, *n1;
    if (l == LL - 1)      { n0 = (float*)d_out; n1 = (float*)d_out + setElems; }
    else if ((l & 1) == 0){ n0 = pA0; n1 = pA1; }
    else                  { n0 = pB0; n1 = pB1; }

    const bool cross = (l & 1) != 0;   // layer types: self, cross, self, cross, ...
    const unsigned short* wql = bWq + (size_t)l * CC * CC;
    const unsigned short* wkl = bWk + (size_t)l * CC * CC;
    const unsigned short* wvl = bWv + (size_t)l * CC * CC;
    const unsigned short* wml = bWm + (size_t)l * CC * CC;
    const unsigned short* w1l = bW1 + (size_t)l * 512 * 512;
    const unsigned short* w2l = bW2 + (size_t)l * 256 * 512;
    const float* bql = bq + l * CC;  const float* bkl = bk + l * CC;
    const float* bvl = bv + l * CC;  const float* bml = bm + l * CC;
    const float* b1l = b1 + l * 512; const float* s1l = s1 + l * 512;
    const float* t1l = t1 + l * 512; const float* b2l = b2 + l * CC;

    for (int s = 0; s < 2; ++s) {
      const float* x   = s ? cur1 : cur0;
      const float* src = cross ? (s ? cur0 : cur1) : x;
      float*       nxt = s ? n1 : n0;

      // Q/K/V projections (q from x; k,v from src)
      gemm_bf16<0><<<gC, blk, 0, stream>>>(wql, x,   nullptr, bql, nullptr, nullptr, nullptr, Qb, 256, 256, 256);
      gemm_bf16<0><<<gC, blk, 0, stream>>>(wkl, src, nullptr, bkl, nullptr, nullptr, nullptr, Kb, 256, 256, 256);
      gemm_bf16<0><<<gC, blk, 0, stream>>>(wvl, src, nullptr, bvl, nullptr, nullptr, nullptr, Vb, 256, 256, 256);
      // scaled-dot-product attention (flash, online softmax)
      flash_attn<<<gA, blk, 0, stream>>>(Qb, Kb, Vb, AO);
      // merge projection
      gemm_bf16<0><<<gC, blk, 0, stream>>>(wml, AO, nullptr, bml, nullptr, nullptr, nullptr, AM, 256, 256, 256);
      // MLP: conv(512<-[x;attn]) + folded-BN affine + ReLU
      gemm_bf16<1><<<gH, blk, 0, stream>>>(w1l, x, AM, b1l, s1l, t1l, nullptr, Hb, 512, 512, 256);
      // conv(256<-512) + residual into next ping buffer (or d_out on last layer)
      gemm_bf16<2><<<gC, blk, 0, stream>>>(w2l, Hb, nullptr, b2l, nullptr, nullptr, x, nxt, 256, 512, 512);
    }
    cur0 = n0;
    cur1 = n1;
  }
}
